// RNN_31456340476000
// MI455X (gfx1250) — matrix operand order repaired and checked by reading, not compiled, not run on hardware
//
#include <hip/hip_runtime.h>

// ---------------------------------------------------------------------------
// LayerNorm-RNN (2 layers) for MI455X / gfx1250, wave32 + WMMA bf16.
//
//  * Weights pre-swizzled fp32 -> bf16 into a v_wmma_f32_16x16x32_bf16
//    B-fragment layout, k = 16*(lane/16) + s like the A loaders (1 KB per 32x16 k-tile x n-tile chunk).
//  * Input x pre-converted to bf16 once (not recurrent).
//  * Persistent kernel: 32 blocks x 256 threads (8 waves). Block owns a
//    32-column slice of H; each wave owns one 16x16 C tile.
//  * B-fragments stream L2 -> LDS via GLOBAL_LOAD_ASYNC_TO_LDS_B128 into a
//    private 4-slot ring per wave (ASYNCcnt-paced, s_wait_asynccnt 6/0),
//    consumed with ds_load_b128. The async stream uses ONE self-incrementing
//    64-bit address pair (v_add_nc_u64 inside the asm) to keep VGPR pressure
//    low. A-fragments use a depth-4 VGPR pipeline with immediate offsets.
//  * Grid-wide sense barrier (L2 atomics + s_sleep): 4 barriers per step.
//  * LayerNorm: deterministic reduction (shfl_xor -> per-(row,ntile) slots
//    -> float4 fold per block -> LDS broadcast).
// ---------------------------------------------------------------------------

#define T_STEPS 512
#define BATCH   64
#define DIN     512
#define HID     1024
#define EPS_LN  1e-5f
#define NBLK    32

typedef __bf16  bf16_t;
typedef bf16_t  v16bf __attribute__((ext_vector_type(16)));
typedef float   v8f   __attribute__((ext_vector_type(8)));
typedef unsigned v8u  __attribute__((ext_vector_type(8)));

__device__ __forceinline__ unsigned pack2_bf16(float a, float b) {
  unsigned ua = __builtin_bit_cast(unsigned, a);
  unsigned ub = __builtin_bit_cast(unsigned, b);
  ua += 0x7fffu + ((ua >> 16) & 1u);
  ub += 0x7fffu + ((ub >> 16) & 1u);
  return (ua >> 16) | (ub & 0xffff0000u);
}
__device__ __forceinline__ unsigned short f2bf(float a) {
  unsigned ua = __builtin_bit_cast(unsigned, a);
  ua += 0x7fffu + ((ua >> 16) & 1u);
  return (unsigned short)(ua >> 16);
}

// ---------------------------------------------------------------------------
// Prep kernels
// ---------------------------------------------------------------------------
__global__ void swizzle_weights(const float* __restrict__ Wx,
                                const float* __restrict__ Wh,
                                int Kx, int KT, unsigned* __restrict__ dst) {
  int idx = blockIdx.x * blockDim.x + threadIdx.x;
  int total = KT * 64 * 32 * 8;
  if (idx >= total) return;
  int p    = idx & 7;
  int lane = (idx >> 3) & 31;
  int ntg  = (idx >> 8) & 63;
  int kt   = idx >> 14;
  int n = ntg * 16 + (lane & 15);
  int k = kt * 32 + p * 2 + ((lane >> 4) * 16);
  float a, b;
  if (k < Kx) { a = Wx[n * Kx + k];   b = Wx[n * Kx + k + 1]; }
  else        { int kk = k - Kx;
                a = Wh[n * HID + kk]; b = Wh[n * HID + kk + 1]; }
  dst[idx] = pack2_bf16(a, b);
}

__global__ void cvt_bf16_bulk(const float* __restrict__ src,
                              unsigned* __restrict__ dst, int npairs) {
  int i = blockIdx.x * blockDim.x + threadIdx.x;
  if (i < npairs) dst[i] = pack2_bf16(src[2 * i], src[2 * i + 1]);
}

__global__ void init_state(const float* __restrict__ hidden,
                           const float* __restrict__ bi0, const float* __restrict__ bh0,
                           const float* __restrict__ bi1, const float* __restrict__ bh1,
                           unsigned short* __restrict__ h0b,
                           unsigned short* __restrict__ h1b,
                           float* __restrict__ bias0, float* __restrict__ bias1,
                           unsigned* __restrict__ bar) {
  int tid = blockIdx.x * blockDim.x + threadIdx.x;
  if (tid < 2 * BATCH * HID) {
    unsigned short v = f2bf(hidden[tid]);
    if (tid < BATCH * HID) h0b[tid] = v;
    else                   h1b[tid - BATCH * HID] = v;
  }
  if (tid < HID) {
    bias0[tid] = bi0[tid] + bh0[tid];
    bias1[tid] = bi1[tid] + bh1[tid];
  }
  if (tid < 2) bar[tid] = 0u;
}

// ---------------------------------------------------------------------------
// Fragment loaders
// ---------------------------------------------------------------------------
__device__ __forceinline__ v16bf load_a_bf16(const unsigned short* __restrict__ A, int ld,
                                             int m0, int k0, int lane) {
  const unsigned short* row = A + (size_t)(m0 + (lane & 15)) * ld + k0 + ((lane >> 4) * 16);
  v8u f;
#pragma unroll
  for (int p = 0; p < 8; ++p) {
    int off = p * 2;
    f[p] = *(const unsigned*)(row + off);   // vectorizes to 2x b128
  }
  return __builtin_bit_cast(v16bf, f);
}
__device__ __forceinline__ v16bf load_a_f32(const float* __restrict__ A, int ld,
                                            int m0, int k0, int lane) {
  const float* row = A + (size_t)(m0 + (lane & 15)) * ld + k0 + ((lane >> 4) * 16);
  v8u f;
#pragma unroll
  for (int p = 0; p < 8; ++p) {
    int off = p * 2;
    f[p] = pack2_bf16(row[off], row[off + 1]);
  }
  return __builtin_bit_cast(v16bf, f);
}
__device__ __forceinline__ v16bf load_b_frag(const unsigned short* __restrict__ wf,
                                             int kt, int ntg, int lane) {
  const v16bf* bp = (const v16bf*)(wf + ((size_t)(kt * 64 + ntg) << 9));
  return bp[lane];
}

// ---------------------------------------------------------------------------
// Async B-fragment staging: L2 -> LDS ring (1 KB fragment = 2 x b128/lane).
// One self-incrementing global address pair per segment (fragment stride is
// 64 ntiles * 1 KB = 0x10000 bytes).
// ---------------------------------------------------------------------------
__device__ __forceinline__ void async_frag_inc(unsigned long long& g, unsigned lds) {
  asm volatile("global_load_async_to_lds_b128 %1, %0, off\n\t"
               "global_load_async_to_lds_b128 %1, %0, off offset:512\n\t"
               "v_add_nc_u64 %0, 0x10000, %0"
               : "+v"(g) : "v"(lds) : "memory");
}
template<int N>
__device__ __forceinline__ void wait_async() {
  asm volatile("s_wait_asynccnt %0" :: "n"(N) : "memory");
}
__device__ __forceinline__ v16bf lds_frag(const unsigned* ring_wave, int slot, int lane) {
  const v16bf* p = (const v16bf*)(ring_wave + slot * 256);
  return p[lane];
}

// Software-pipelined GEMM segment: B via async LDS ring, A depth-4 pipeline.
template<int NKT>
__device__ __forceinline__ void gemm_seg_async(v8f& acc,
                                               const unsigned short* __restrict__ Asrc, int ld,
                                               int m0,
                                               const unsigned short* __restrict__ wf,
                                               int ktW0, int ntg, int lane,
                                               unsigned ring_base, const unsigned* ring_wave) {
  static_assert(NKT >= 5, "segment too short");
  const unsigned lbase = ring_base + (unsigned)(lane * 16);
  unsigned long long g = (unsigned long long)((const char*)wf +
                         ((size_t)(ktW0 * 64 + ntg) << 10)) + (unsigned)(lane * 16);
  // prologue: stage fragments 0,1,2 ; preload A(0..3)
  async_frag_inc(g, lbase + 0 * 1024);
  async_frag_inc(g, lbase + 1 * 1024);
  async_frag_inc(g, lbase + 2 * 1024);
  v16bf areg[4];
#pragma unroll
  for (int i = 0; i < 4; ++i) areg[i] = load_a_bf16(Asrc, ld, m0, i * 32, lane);

#pragma unroll
  for (int kt = 0; kt <= NKT - 5; ++kt) {
    async_frag_inc(g, lbase + (unsigned)(((kt + 3) & 3) << 10));   // frag kt+3
    v16bf an = load_a_bf16(Asrc, ld, m0, (kt + 4) * 32, lane);
    wait_async<6>();                         // frag kt complete; 3 in flight
    v16bf b = lds_frag(ring_wave, kt & 3, lane);
    acc = __builtin_amdgcn_wmma_f32_16x16x32_bf16(false, areg[kt & 3], false, b,
                                                  (short)0, acc, false, false);
    areg[kt & 3] = an;
  }
  // kt = NKT-4: last issue (frag NKT-1), consume frag NKT-4
  {
    async_frag_inc(g, lbase + (unsigned)(((NKT - 1) & 3) << 10));
    wait_async<6>();
    v16bf b = lds_frag(ring_wave, (NKT - 4) & 3, lane);
    acc = __builtin_amdgcn_wmma_f32_16x16x32_bf16(false, areg[(NKT - 4) & 3], false, b,
                                                  (short)0, acc, false, false);
  }
  wait_async<0>();                           // ring fully populated
#pragma unroll
  for (int j = NKT - 3; j < NKT; ++j) {
    v16bf b = lds_frag(ring_wave, j & 3, lane);
    acc = __builtin_amdgcn_wmma_f32_16x16x32_bf16(false, areg[j & 3], false, b,
                                                  (short)0, acc, false, false);
  }
}

// fallback segment (fp32 A converted inline; B direct from L2)
__device__ __forceinline__ void gemm_seg_f32(v8f& acc,
                                             const float* __restrict__ Asrc, int ld,
                                             int m0, int nkt,
                                             const unsigned short* __restrict__ wf,
                                             int ktW0, int ntg, int lane) {
  v16bf a = load_a_f32(Asrc, ld, m0, 0, lane);
  v16bf b = load_b_frag(wf, ktW0, ntg, lane);
#pragma unroll 2
  for (int i = 0; i < nkt - 1; ++i) {
    v16bf an = load_a_f32(Asrc, ld, m0, (i + 1) * 32, lane);
    v16bf bn = load_b_frag(wf, ktW0 + i + 1, ntg, lane);
    acc = __builtin_amdgcn_wmma_f32_16x16x32_bf16(false, a, false, b,
                                                  (short)0, acc, false, false);
    a = an; b = bn;
  }
  acc = __builtin_amdgcn_wmma_f32_16x16x32_bf16(false, a, false, b,
                                                (short)0, acc, false, false);
}

// grid-wide sense barrier
__device__ __forceinline__ void grid_barrier(unsigned* cnt, unsigned* gen, unsigned nb) {
  __syncthreads();
  if (threadIdx.x == 0) {
    __threadfence();
    unsigned g = atomicAdd(gen, 0u);
    if (atomicAdd(cnt, 1u) == nb - 1u) {
      atomicExch(cnt, 0u);
      __threadfence();
      atomicAdd(gen, 1u);
    } else {
      while (atomicAdd(gen, 0u) == g) { __builtin_amdgcn_s_sleep(2); }
    }
    __threadfence();
  }
  __syncthreads();
}

// ---------------------------------------------------------------------------
// Persistent RNN kernel.
// ---------------------------------------------------------------------------
__global__ __launch_bounds__(256)
void rnn_persistent(const float* __restrict__ x,            // [T,B,DIN] fp32
                    const unsigned short* __restrict__ xbf, // [T,B,DIN] bf16 or null
                    const float* __restrict__ lnw0, const float* __restrict__ lnb0,
                    const float* __restrict__ lnw1, const float* __restrict__ lnb1,
                    const unsigned short* __restrict__ wf0,
                    const unsigned short* __restrict__ wf1,
                    unsigned short* __restrict__ h0b,
                    unsigned short* __restrict__ h1b,
                    const float* __restrict__ bias0, const float* __restrict__ bias1,
                    float* __restrict__ part,               // [2][2][64][64]
                    unsigned* __restrict__ bar,
                    float* __restrict__ out) {              // [T,B,HID] + [2,B,HID]
  __shared__ unsigned ringmem[8 * 4 * 256];   // 8 waves x 4 slots x 1 KB = 32 KB
  __shared__ float smu[BATCH];
  __shared__ float srs[BATCH];

  const int tid   = threadIdx.x;
  const int lane  = tid & 31;
  const int wid   = tid >> 5;
  const int m0    = (wid & 3) * 16;
  const int ntile = wid >> 2;
  const int ntg   = blockIdx.x * 2 + ntile;
  const int col   = ntg * 16 + (lane & 15);
  const int half  = lane >> 4;
  const unsigned nb = gridDim.x;

  const unsigned* ring_wave = ringmem + wid * 1024;
  const unsigned  ring_base = (unsigned)(size_t)(const void*)ring_wave;

  float* P0 = part;
  float* Q0 = part + 64 * 64;
  float* P1 = part + 2 * 64 * 64;
  float* Q1 = part + 3 * 64 * 64;

  const float lw0 = lnw0[col], lb0 = lnb0[col];
  const float lw1 = lnw1[col], lb1 = lnb1[col];
  const float bv0 = bias0[col], bv1 = bias1[col];

#pragma unroll 1
  for (int t = 0; t < T_STEPS; ++t) {
    // ===== layer 0: pre = x_t @ Wih0^T + h0 @ Whh0^T, K = 512 + 1024 =======
    v8f acc = {0.f, 0.f, 0.f, 0.f, 0.f, 0.f, 0.f, 0.f};
    if (xbf) {
      gemm_seg_async<16>(acc, xbf + (size_t)t * BATCH * DIN, DIN, m0,
                         wf0, 0, ntg, lane, ring_base, ring_wave);
    } else {
      gemm_seg_f32(acc, x + (size_t)t * BATCH * DIN, DIN, m0, 16, wf0, 0, ntg, lane);
    }
    gemm_seg_async<32>(acc, h0b, HID, m0, wf0, 16, ntg, lane, ring_base, ring_wave);

#pragma unroll
    for (int r = 0; r < 8; ++r) {
      float v = acc[r] + bv0;
      acc[r] = v;
      float q = v * v;
#pragma unroll
      for (int m = 8; m >= 1; m >>= 1) {
        v += __shfl_xor(v, m, 16);
        q += __shfl_xor(q, m, 16);
      }
      if ((lane & 15) == 0) {
        int row = m0 + r + half * 8;
        P0[row * 64 + ntg] = v;
        Q0[row * 64 + ntg] = q;
      }
    }
    grid_barrier(bar, bar + 1, nb);
    if (tid < BATCH) {
      const float4* ps = (const float4*)(P0 + tid * 64);
      const float4* pq = (const float4*)(Q0 + tid * 64);
      float s = 0.f, q = 0.f;
#pragma unroll 4
      for (int i = 0; i < 16; ++i) {
        float4 a = ps[i], c = pq[i];
        s += (a.x + a.y) + (a.z + a.w);
        q += (c.x + c.y) + (c.z + c.w);
      }
      float mu  = s * (1.f / (float)HID);
      float var = q * (1.f / (float)HID) - mu * mu;
      smu[tid] = mu;
      srs[tid] = __frsqrt_rn(var + EPS_LN);
    }
    __syncthreads();
#pragma unroll
    for (int r = 0; r < 8; ++r) {
      int row = m0 + r + half * 8;
      float y = (acc[r] - smu[row]) * srs[row] * lw0 + lb0;
      float h = tanhf(y);
      h0b[row * HID + col] = f2bf(h);
      if (t == T_STEPS - 1)
        out[(size_t)T_STEPS * BATCH * HID + (size_t)row * HID + col] = h;
    }
    grid_barrier(bar, bar + 1, nb);

    // ===== layer 1: pre = h0 @ Wih1^T + h1 @ Whh1^T, K = 1024 + 1024 =======
    v8f acc1 = {0.f, 0.f, 0.f, 0.f, 0.f, 0.f, 0.f, 0.f};
    gemm_seg_async<32>(acc1, h0b, HID, m0, wf1, 0,  ntg, lane, ring_base, ring_wave);
    gemm_seg_async<32>(acc1, h1b, HID, m0, wf1, 32, ntg, lane, ring_base, ring_wave);

#pragma unroll
    for (int r = 0; r < 8; ++r) {
      float v = acc1[r] + bv1;
      acc1[r] = v;
      float q = v * v;
#pragma unroll
      for (int m = 8; m >= 1; m >>= 1) {
        v += __shfl_xor(v, m, 16);
        q += __shfl_xor(q, m, 16);
      }
      if ((lane & 15) == 0) {
        int row = m0 + r + half * 8;
        P1[row * 64 + ntg] = v;
        Q1[row * 64 + ntg] = q;
      }
    }
    grid_barrier(bar, bar + 1, nb);
    if (tid < BATCH) {
      const float4* ps = (const float4*)(P1 + tid * 64);
      const float4* pq = (const float4*)(Q1 + tid * 64);
      float s = 0.f, q = 0.f;
#pragma unroll 4
      for (int i = 0; i < 16; ++i) {
        float4 a = ps[i], c = pq[i];
        s += (a.x + a.y) + (a.z + a.w);
        q += (c.x + c.y) + (c.z + c.w);
      }
      float mu  = s * (1.f / (float)HID);
      float var = q * (1.f / (float)HID) - mu * mu;
      smu[tid] = mu;
      srs[tid] = __frsqrt_rn(var + EPS_LN);
    }
    __syncthreads();
#pragma unroll
    for (int r = 0; r < 8; ++r) {
      int row = m0 + r + half * 8;
      float y = (acc1[r] - smu[row]) * srs[row] * lw1 + lb1;
      float h = tanhf(y);
      h1b[row * HID + col] = f2bf(h);
      out[(size_t)t * BATCH * HID + (size_t)row * HID + col] = h;
      if (t == T_STEPS - 1)
        out[(size_t)(T_STEPS + 1) * BATCH * HID + (size_t)row * HID + col] = h;
    }
    grid_barrier(bar, bar + 1, nb);
  }
}

// ---------------------------------------------------------------------------
extern "C" void kernel_launch(void* const* d_in, const int* in_sizes, int n_in,
                              void* d_out, int out_size, void* d_ws, size_t ws_size,
                              hipStream_t stream) {
  const float* input  = (const float*)d_in[0];
  const float* hidden = (const float*)d_in[1];
  const float* W_ih0  = (const float*)d_in[2];
  const float* W_hh0  = (const float*)d_in[3];
  const float* b_ih0  = (const float*)d_in[4];
  const float* b_hh0  = (const float*)d_in[5];
  const float* ln_w0  = (const float*)d_in[6];
  const float* ln_b0  = (const float*)d_in[7];
  const float* W_ih1  = (const float*)d_in[8];
  const float* W_hh1  = (const float*)d_in[9];
  const float* b_ih1  = (const float*)d_in[10];
  const float* b_hh1  = (const float*)d_in[11];
  const float* ln_w1  = (const float*)d_in[12];
  const float* ln_b1  = (const float*)d_in[13];

  char* w = (char*)d_ws;
  unsigned short* wf0 = (unsigned short*)(w);            // 3,145,728 B
  unsigned short* wf1 = (unsigned short*)(w + 3145728);  // 4,194,304 B
  unsigned short* h0b = (unsigned short*)(w + 7340032);  // 131,072 B
  unsigned short* h1b = (unsigned short*)(w + 7471104);  // 131,072 B
  float*          bias0 = (float*)(w + 7602176);         // 4,096 B
  float*          bias1 = (float*)(w + 7606272);         // 4,096 B
  float*          part  = (float*)(w + 7610368);         // 65,536 B
  unsigned*       bar   = (unsigned*)(w + 7675904);      // 256 B
  unsigned short* xbf   = (unsigned short*)(w + 7676160);// 33,554,432 B (optional)
  const size_t WS_NEEDED_XBF = 7676160u + (size_t)T_STEPS * BATCH * DIN * 2u;
  const bool use_xbf = (ws_size >= WS_NEEDED_XBF);
  float* out = (float*)d_out;

  swizzle_weights<<<(48 * 64 * 32 * 8) / 256, 256, 0, stream>>>(
      W_ih0, W_hh0, 512, 48, (unsigned*)wf0);
  swizzle_weights<<<(64 * 64 * 32 * 8) / 256, 256, 0, stream>>>(
      W_ih1, W_hh1, 1024, 64, (unsigned*)wf1);
  init_state<<<512, 256, 0, stream>>>(hidden, b_ih0, b_hh0, b_ih1, b_hh1,
                                      h0b, h1b, bias0, bias1, bar);
  if (use_xbf) {
    int npairs = T_STEPS * BATCH * DIN / 2;               // 8,388,608
    cvt_bf16_bulk<<<npairs / 256, 256, 0, stream>>>(input, (unsigned*)xbf, npairs);
  }
  rnn_persistent<<<NBLK, 256, 0, stream>>>(input, use_xbf ? xbf : (unsigned short*)nullptr,
                                           ln_w0, ln_b0, ln_w1, ln_b1,
                                           wf0, wf1, h0b, h1b, bias0, bias1,
                                           part, bar, out);
}
